// Head_41077067219530
// MI455X (gfx1250) — matrix operand order repaired and checked by reading, not compiled, not run on hardware
//
#include <hip/hip_runtime.h>
#include <stdint.h>

typedef __attribute__((ext_vector_type(16))) __bf16 v16bf;
typedef __attribute__((ext_vector_type(8)))  float  v8f;
typedef __attribute__((ext_vector_type(4)))  unsigned int v4u;
typedef __attribute__((ext_vector_type(8)))  int v8i_;
typedef __attribute__((ext_vector_type(4)))  int v4i_;

#define BATCH 8
#define SEQ   2048
#define DEMB  1024
#define HS    64
#define NROWS (BATCH * SEQ)
#define SPLIT 4
#define KV_PER_SPLIT (SEQ / SPLIT)   // 512

// A-fragment (16-bit, 16xK): lane holds row M=lane%16; VGPR j / half h holds
// K = 8g+2j+h (j<4) or 16+8g+2(j-4)+h (j>=4), g = lane/16.  (ISA 7.12.2)
__device__ __forceinline__ int koffA(int j, int g) {
  return (j < 4) ? (8 * g + 2 * j) : (16 + 8 * g + 2 * (j - 4));
}
// B-fragment (16-bit, Kx16): lane holds col N=lane%16; VGPR j / half h holds
// K = 8g+2j+h (j<4) or 16+8g+2(j-4)+h (j>=4), g = lane/16 (matches A).
__device__ __forceinline__ int koffB(int j, int g) {
  return (j < 4) ? (8 * g + 2 * j) : (16 + 8 * g + 2 * (j - 4));
}

__device__ __forceinline__ v8f wmma_bf16(v16bf a, v16bf b, v8f c) {
  return __builtin_amdgcn_wmma_f32_16x16x32_bf16(
      false, a, false, b, (short)0, c, false, false);
}

__device__ __forceinline__ uint32_t lds_off(const void* p) {
  // generic pointer to LDS: addr[31:0] is the LDS byte offset (ISA 10.2)
  return (uint32_t)(uintptr_t)p;
}

// ---------------------------------------------------------------------------
// Tensor Data Mover: issue a 1D/2D tile load (global -> LDS).
// D# group0/group1 packed per ISA 8.3 / 8.4.  data_size = 8B units (code 3).
// tile_d1 == 0 -> 1D tile.  Issued by one wave; tracked via TENSORcnt.
// amdgpu-toolchain (clang-23) builtin arity: (v4u, v8i, v4i, v4i, v8i, cpol).
// ---------------------------------------------------------------------------
__device__ __forceinline__ void tdm_load(uint32_t lds, const void* gptr,
                                         uint32_t tile_d0, uint32_t tile_d1,
                                         uint32_t tens_d0, uint32_t tens_d1,
                                         uint32_t stride0) {
  const uint64_t ga = (uint64_t)(uintptr_t)gptr;
  v4u g0;
  g0[0] = 1u;                                   // count=1, user descriptor
  g0[1] = lds;                                  // lds_addr (bytes)
  g0[2] = (uint32_t)ga;                         // global_addr[31:0]
  g0[3] = (uint32_t)((ga >> 32) & 0x01FFFFFFu)  // global_addr[56:32]
          | (2u << 30);                         // type = 2 ("image")
  v8i_ g1;
  g1[0] = (int)(3u << 16);                      // data_size = 8 bytes
  g1[1] = (int)((tens_d0 & 0xFFFFu) << 16);     // tensor_dim0[15:0]
  g1[2] = (int)(((tens_d0 >> 16) & 0xFFFFu) | ((tens_d1 & 0xFFFFu) << 16));
  g1[3] = (int)(((tens_d1 >> 16) & 0xFFFFu) | ((tile_d0 & 0xFFFFu) << 16));
  g1[4] = (int)(tile_d1 & 0xFFFFu);             // tile_dim1 (0 -> 1D), tile_dim2=0
  g1[5] = (int)stride0;                         // tensor_dim0_stride[31:0]
  g1[6] = 0;                                    // stride0[47:32]=0, dim1_stride lo=0
  g1[7] = 0;
  __builtin_amdgcn_tensor_load_to_lds(g0, g1, (v4i_)0, (v4i_)0, (v8i_)0, 0);
}

// ---------------------------------------------------------------------------
// Kernel 0: convert W -> bf16 and transpose into WtG[m][col][d] so TDM can
// stage contiguous 64B rows per output column.
// ---------------------------------------------------------------------------
__global__ __launch_bounds__(256) void prep_w(
    const float* __restrict__ Wq, const float* __restrict__ Wk,
    const float* __restrict__ Wv, __bf16* __restrict__ WtG) {
  const int idx = blockIdx.x * 256 + threadIdx.x;   // over 3*DEMB*HS
  const int m   = idx / (DEMB * HS);
  const int r   = idx - m * (DEMB * HS);
  const int d   = r >> 6;          // coalesced read: consecutive tid -> col
  const int col = r & (HS - 1);
  const float* W = (m == 0) ? Wq : (m == 1) ? Wk : Wv;
  WtG[((size_t)m * HS + col) * DEMB + d] = (__bf16)W[d * HS + col];
}

// ---------------------------------------------------------------------------
// Kernel 1: fused Q/K/V projection.  One wave per 16-row tile of x (read once,
// fp32->bf16 into A fragments); W slices double-buffered in LDS via TDM.
// Writes Q,K row-major bf16 and V transposed (VbT[b][hd][s]) for kernel 2.
// ---------------------------------------------------------------------------
__global__ __launch_bounds__(128) void qkv_project(
    const float* __restrict__ x, const __bf16* __restrict__ WtG,
    __bf16* __restrict__ Qb, __bf16* __restrict__ Kb, __bf16* __restrict__ VbT)
{
  __shared__ __bf16 Wt[2][3][HS][32];   // [buf][matrix][out-col][k in slice]

  const int tid  = threadIdx.x;
  const int wave = tid >> 5;
  const int lane = tid & 31;
  const int g    = lane >> 4;
  const int lm   = lane & 15;
  const bool wu0 = (__builtin_amdgcn_readfirstlane(tid) >> 5) == 0;

  const long r0  = ((long)blockIdx.x * 4 + wave) * 16;
  const long row = r0 + lm;

  // TDM: 2D tile per matrix: 64 rows (out-col) x 64B (32 k's), row stride 2KB
  auto issue_w = [&](int s, int buf) {
#pragma unroll
    for (int m = 0; m < 3; ++m)
      tdm_load(lds_off(&Wt[buf][m][0][0]),
               WtG + (size_t)m * HS * DEMB + s * 32,
               /*tile_d0=*/8, /*tile_d1=*/HS,
               /*tens_d0=*/DEMB / 4, /*tens_d1=*/HS,
               /*stride0=*/DEMB / 4);
  };

  v8f acc[3][4] = {};
  if (wu0) issue_w(0, 0);

  const int nsteps = DEMB / 32;
  for (int s = 0; s < nsteps; ++s) {
    __builtin_amdgcn_s_wait_tensorcnt(0);
    __syncthreads();                       // tile s in LDS for all waves
    if (wu0 && s + 1 < nsteps) issue_w(s + 1, (s + 1) & 1);
    const int buf = s & 1;
    const int kb  = s * 32;

    v16bf a;
#pragma unroll
    for (int j = 0; j < 8; ++j) {
      const int ko   = koffA(j, g);
      const float2 f = *(const float2*)(x + row * DEMB + kb + ko);
      a[2 * j]     = (__bf16)f.x;
      a[2 * j + 1] = (__bf16)f.y;
    }

#pragma unroll
    for (int m = 0; m < 3; ++m)
#pragma unroll
      for (int n = 0; n < 4; ++n) {
        v16bf bf;
        const int col = n * 16 + lm;
#pragma unroll
        for (int j = 0; j < 8; ++j) {
          const int ko = koffB(j, g);
          bf[2 * j]     = Wt[buf][m][col][ko];
          bf[2 * j + 1] = Wt[buf][m][col][ko + 1];
        }
        acc[m][n] = wmma_bf16(a, bf, acc[m][n]);
      }
  }

#pragma unroll
  for (int n = 0; n < 4; ++n)
#pragma unroll
    for (int r = 0; r < 8; ++r) {
      const long grow = r0 + r + 8 * g;
      const int  hd   = n * 16 + lm;
      Qb[grow * HS + hd] = (__bf16)acc[0][n][r];
      Kb[grow * HS + hd] = (__bf16)acc[1][n][r];
      const long bb = grow >> 11, srow = grow & (SEQ - 1);
      VbT[(bb * HS + hd) * SEQ + srow] = (__bf16)acc[2][n][r];
    }
}

// ---------------------------------------------------------------------------
// Kernel 2: flash attention, split-K over 4 chunks of 512 keys.  8 waves share
// TDM-staged, double-buffered 32-key K/V tiles; online softmax in f32; writes
// unnormalized partial O plus (m, l) per row.
// ---------------------------------------------------------------------------
__global__ __launch_bounds__(256) void flash_attn_partial(
    const __bf16* __restrict__ Qb, const __bf16* __restrict__ Kb,
    const __bf16* __restrict__ VbT, float* __restrict__ Opart,
    float2* __restrict__ ML)
{
  __shared__ __bf16 Kt[2][32][HS];      // [buf][key][head-dim]
  __shared__ __bf16 Vt[2][HS][32];      // [buf][head-dim][key]
  __shared__ __bf16 Psc[8][16][32];     // per-wave probability scratch

  const int tid  = threadIdx.x;
  const int wave = tid >> 5;
  const int lane = tid & 31;
  const int g    = lane >> 4;
  const int lm   = lane & 15;
  const bool wu0 = (__builtin_amdgcn_readfirstlane(tid) >> 5) == 0;

  const int  split = blockIdx.x & (SPLIT - 1);
  const int  qc    = blockIdx.x >> 2;
  const int  b     = qc >> 4;
  const int  chunk = qc & 15;
  const long q0    = (long)b * SEQ + chunk * 128 + wave * 16;
  const int  kv0   = split * KV_PER_SPLIT;

  auto issue_kv = [&](int i, int buf) {
    const int kv = kv0 + i * 32;
    // K tile: 32 rows x 64 hd, fully contiguous -> 1D 4KB copy
    tdm_load(lds_off(&Kt[buf][0][0]),
             Kb + ((size_t)b * SEQ + kv) * HS,
             /*tile_d0=*/512, /*tile_d1=*/0,
             /*tens_d0=*/1u << 20, /*tens_d1=*/0, /*stride0=*/512);
    // V tile: 64 rows (hd) x 64B (32 keys), row stride SEQ*2B
    tdm_load(lds_off(&Vt[buf][0][0]),
             VbT + (size_t)b * HS * SEQ + kv,
             /*tile_d0=*/8, /*tile_d1=*/HS,
             /*tens_d0=*/SEQ / 4, /*tens_d1=*/HS,
             /*stride0=*/SEQ / 4);
  };

  // Q fragments: loaded once, reused for all key tiles
  v16bf qa[2];
#pragma unroll
  for (int t = 0; t < 2; ++t)
#pragma unroll
    for (int j = 0; j < 8; ++j) {
      const int ko = koffA(j, g);
      qa[t][2 * j]     = Qb[(q0 + lm) * HS + 32 * t + ko];
      qa[t][2 * j + 1] = Qb[(q0 + lm) * HS + 32 * t + ko + 1];
    }

  v8f accO[4] = {};
  float mrow[8], lrow[8];
#pragma unroll
  for (int r = 0; r < 8; ++r) { mrow[r] = -3.0e38f; lrow[r] = 0.0f; }

  const float scale = 0.125f;   // HEAD_SIZE^-0.5

  if (wu0) issue_kv(0, 0);
  const int nsteps = KV_PER_SPLIT / 32;
  for (int i = 0; i < nsteps; ++i) {
    __builtin_amdgcn_s_wait_tensorcnt(0);
    __syncthreads();                         // tile i visible to all waves
    if (wu0 && i + 1 < nsteps) issue_kv(i + 1, (i + 1) & 1);
    const int buf = i & 1;

    // scores: S(16x32) = Q(16x64) x K^T, 2 key sub-tiles x 2 head-dim steps
    v8f st[2] = {};
#pragma unroll
    for (int t = 0; t < 2; ++t)
#pragma unroll
      for (int s = 0; s < 2; ++s) {
        v16bf bk;
        const int key = s * 16 + lm;
#pragma unroll
        for (int j = 0; j < 8; ++j) {
          const int ko = koffB(j, g);
          bk[2 * j]     = Kt[buf][key][32 * t + ko];
          bk[2 * j + 1] = Kt[buf][key][32 * t + ko + 1];
        }
        st[s] = wmma_bf16(qa[t], bk, st[s]);
      }

    // online softmax over 32 new columns (rows r+8g live in lane-half g)
#pragma unroll
    for (int r = 0; r < 8; ++r) {
      float s0 = st[0][r] * scale;
      float s1 = st[1][r] * scale;
      float tm = fmaxf(s0, s1);
      tm = fmaxf(tm, __shfl_xor(tm, 8, 32));
      tm = fmaxf(tm, __shfl_xor(tm, 4, 32));
      tm = fmaxf(tm, __shfl_xor(tm, 2, 32));
      tm = fmaxf(tm, __shfl_xor(tm, 1, 32));
      const float mnew  = fmaxf(mrow[r], tm);
      const float alpha = __expf(mrow[r] - mnew);
      const float p0 = __expf(s0 - mnew);
      const float p1 = __expf(s1 - mnew);
      float rs = p0 + p1;
      rs += __shfl_xor(rs, 8, 32);
      rs += __shfl_xor(rs, 4, 32);
      rs += __shfl_xor(rs, 2, 32);
      rs += __shfl_xor(rs, 1, 32);
      lrow[r] = lrow[r] * alpha + rs;
      mrow[r] = mnew;
#pragma unroll
      for (int n = 0; n < 4; ++n) accO[n][r] *= alpha;
      const int rowl = r + 8 * g;
      Psc[wave][rowl][lm]      = (__bf16)p0;
      Psc[wave][rowl][16 + lm] = (__bf16)p1;
    }

    // P fragment (16x32, A-layout) for the PV matmul
    v16bf pf;
#pragma unroll
    for (int j = 0; j < 8; ++j) {
      const int ko = koffA(j, g);
      pf[2 * j]     = Psc[wave][lm][ko];
      pf[2 * j + 1] = Psc[wave][lm][ko + 1];
    }

#pragma unroll
    for (int n = 0; n < 4; ++n) {
      v16bf bv;
      const int hd = n * 16 + lm;
#pragma unroll
      for (int j = 0; j < 8; ++j) {
        const int ko = koffB(j, g);
        bv[2 * j]     = Vt[buf][hd][ko];
        bv[2 * j + 1] = Vt[buf][hd][ko + 1];
      }
      accO[n] = wmma_bf16(pf, bv, accO[n]);
    }
  }

  // partial outputs: unnormalized O plus per-row (m, l)
#pragma unroll
  for (int r = 0; r < 8; ++r) {
    const long grow = q0 + r + 8 * g;
    const size_t base = ((size_t)split * NROWS + grow) * HS;
#pragma unroll
    for (int n = 0; n < 4; ++n)
      Opart[base + n * 16 + lm] = accO[n][r];
    if (lm == 0)
      ML[(size_t)split * NROWS + grow] = make_float2(mrow[r], lrow[r]);
  }
}

// ---------------------------------------------------------------------------
// Kernel 3: merge the SPLIT partials: out = sum_s e^(m_s-m*) O_s / sum_s e^(m_s-m*) l_s
// ---------------------------------------------------------------------------
__global__ __launch_bounds__(256) void attn_combine(
    const float* __restrict__ Opart, const float2* __restrict__ ML,
    float* __restrict__ out)
{
  const size_t idx = (size_t)blockIdx.x * 256 + threadIdx.x;  // NROWS*HS
  const size_t row = idx >> 6;
  const int    hd  = (int)(idx & (HS - 1));
  float2 ml[SPLIT];
  float mstar = -3.0e38f;
#pragma unroll
  for (int s = 0; s < SPLIT; ++s) {
    ml[s] = ML[(size_t)s * NROWS + row];
    mstar = fmaxf(mstar, ml[s].x);
  }
  float acc = 0.0f, den = 0.0f;
#pragma unroll
  for (int s = 0; s < SPLIT; ++s) {
    const float w = __expf(ml[s].x - mstar);
    acc += w * Opart[((size_t)s * NROWS + row) * HS + hd];
    den += w * ml[s].y;
  }
  out[idx] = acc / den;
}

// ---------------------------------------------------------------------------
extern "C" void kernel_launch(void* const* d_in, const int* in_sizes, int n_in,
                              void* d_out, int out_size, void* d_ws, size_t ws_size,
                              hipStream_t stream) {
  (void)in_sizes; (void)n_in; (void)out_size; (void)ws_size;
  const float* x  = (const float*)d_in[0];
  const float* Wq = (const float*)d_in[1];
  const float* Wk = (const float*)d_in[2];
  const float* Wv = (const float*)d_in[3];

  char* ws = (char*)d_ws;
  __bf16* Qb   = (__bf16*)(ws);                                   // 2 MB
  __bf16* Kb   = (__bf16*)(ws + (size_t)2 * 1024 * 1024);         // 2 MB
  __bf16* VbT  = (__bf16*)(ws + (size_t)4 * 1024 * 1024);         // 2 MB
  __bf16* WtG  = (__bf16*)(ws + (size_t)6 * 1024 * 1024);         // 384 KB
  float*  Opart = (float*)(ws + (size_t)8 * 1024 * 1024);         // 16 MB
  float2* ML    = (float2*)(ws + (size_t)24 * 1024 * 1024);       // 512 KB

  prep_w<<<(3 * DEMB * HS) / 256, 256, 0, stream>>>(Wq, Wk, Wv, WtG);
  qkv_project<<<NROWS / 16 / 4, 128, 0, stream>>>(x, WtG, Qb, Kb, VbT);
  flash_attn_partial<<<(NROWS / 128) * SPLIT, 256, 0, stream>>>(Qb, Kb, VbT, Opart, ML);
  attn_combine<<<(NROWS * HS) / 256, 256, 0, stream>>>(Opart, ML, (float*)d_out);
}
